// MHAttention_63084479643897
// MI455X (gfx1250) — matrix operand order repaired and checked by reading, not compiled, not run on hardware
//
#include <hip/hip_runtime.h>
#include <hip/hip_bf16.h>
#include <math.h>

// Problem constants (match reference).
#define BB   32
#define SS   2048
#define QD   1024
#define KVD  1024
#define NH   8
#define HD   128
#define TS   64        // S-tile rows per iteration
#define LDP  136       // padded LDS row stride (bf16 elems) to avoid bank conflicts

typedef __attribute__((ext_vector_type(16))) __bf16 v16bf;
typedef __attribute__((ext_vector_type(8)))  __bf16 v8bf;
typedef __attribute__((ext_vector_type(4)))  __bf16 v4bf;
typedef __attribute__((ext_vector_type(8)))  float  v8f;

// ---- WMMA bf16 fragment loaders (16x16x32, wave32 layouts per ISA 7.12.2) ----
// A (16x32 bf16): lane m=lane&15; grp=lane>>4; elems 0..7 -> K=grp*8+e,
//                 elems 8..15 -> K=16+grp*8+(e-8)  => two contiguous 8-elem runs.
__device__ __forceinline__ v16bf ld_frag_a(const __bf16* rowp, int kb, int grp) {
    const v8bf lo = *(const v8bf*)(rowp + kb + grp * 8);
    const v8bf hi = *(const v8bf*)(rowp + kb + 16 + grp * 8);
    return __builtin_shufflevector(lo, hi, 0,1,2,3,4,5,6,7,8,9,10,11,12,13,14,15);
}
// B (32x16 bf16): lane n=lane&15 selects column-row; elems 0..7 -> K=grp*8+e, 8..15 -> K=16+grp*8+(e-8).
__device__ __forceinline__ v16bf ld_frag_b(const __bf16* rowp, int kb, int grp) {
    const v8bf lo = *(const v8bf*)(rowp + kb + grp * 8);
    const v8bf hi = *(const v8bf*)(rowp + kb + 16 + grp * 8);
    return __builtin_shufflevector(lo, hi, 0,1,2,3,4,5,6,7,8,9,10,11,12,13,14,15);
}

// ---------------------------------------------------------------------------
// Kernel 0: pack Ww (8x128x1024) and Wc (8x128x128) fp32 -> bf16 in workspace.
// Halves L2 restream traffic for the weight operands of both GEMMs.
// ---------------------------------------------------------------------------
__global__ __launch_bounds__(256)
void prepack_kernel(const float* __restrict__ Ww, const float* __restrict__ Wc,
                    unsigned short* __restrict__ WwB_us, unsigned short* __restrict__ WcB_us) {
    __bf16* WwB = (__bf16*)WwB_us;
    __bf16* WcB = (__bf16*)WcB_us;
    const int nWw = NH * HD * KVD;          // 1048576
    const int nWc = NH * HD * HD;           // 131072
    for (int i = blockIdx.x * blockDim.x + threadIdx.x; i < nWw + nWc;
         i += gridDim.x * blockDim.x) {
        if (i < nWw) WwB[i] = (__bf16)Ww[i];
        else         WcB[i - nWw] = (__bf16)Wc[i - nWw];
    }
}

// ---------------------------------------------------------------------------
// Kernel 1: fused per-(b,h) pipeline.
//   head_kv tile (WMMA bf16, f32 acc) -> uh (WMMA bf16) -> tanh scores
//   -> online softmax (m,l) + f32 ctx accumulation (flash-style).
// Writes: ctx (d_out), raw scores + m,l (workspace) for pass 2.
// ---------------------------------------------------------------------------
__global__ __launch_bounds__(256)
void fused_attn_kernel(const float* __restrict__ q,
                       const float* __restrict__ kv,
                       const int*   __restrict__ kvmask,
                       const float* __restrict__ bw,
                       const float* __restrict__ Wq,
                       const float* __restrict__ bq,
                       const float* __restrict__ Wv,
                       const float* __restrict__ bv,
                       const unsigned short* __restrict__ WwB_us,
                       const unsigned short* __restrict__ WcB_us,
                       float* __restrict__ scoresW,
                       float* __restrict__ mW,
                       float* __restrict__ lW,
                       float* __restrict__ outCtx) {
    __shared__ __bf16 smAB[TS * LDP];     // kv chunk (GEMM1 A) / head_kv bf16 (GEMM2 A)
    __shared__ __bf16 smW[HD * LDP];      // Ww k-chunk (GEMM1 B) / Wc (GEMM2 B)
    __shared__ float  sm_wq[HD];
    __shared__ float  sm_bw[HD];
    __shared__ float  sm_wv[HD];
    __shared__ float  sm_c[HD];
    __shared__ float  sm_score[TS];
    __shared__ float  sm_ml[4];           // [0]=m, [1]=l, [2]=new_m, [3]=scale

    const __bf16* WwB = (const __bf16*)WwB_us;
    const __bf16* WcB = (const __bf16*)WcB_us;

    const int bh = blockIdx.x;
    const int b  = bh / NH;
    const int h  = bh % NH;

    const int tid  = threadIdx.x;
    const int lane = tid & 31;
    const int wave = tid >> 5;            // 0..7
    const int grp  = lane >> 4;           // 0/1
    const int ln   = lane & 15;
    const int mb   = wave & 3;            // 16-row block within 64-row tile
    const int cb   = (wave >> 2) * 64;    // 64-col half
    const int arow = mb * 16 + ln;        // A-fragment row for this lane

    const float bv_h = bv[h];
    const int*  maskp = kvmask + (size_t)b * SS;
    const __bf16* wsrcBase = WwB + (size_t)h * HD * KVD;
    const __bf16* csrc     = WcB + (size_t)h * HD * HD;

    // ---- per-block init: wq = Wq[h] q[b] + bq ; stage bw, Wv ; zero state ----
    if (tid < HD) {
        const float4* qp = (const float4*)(q + (size_t)b * QD);
        const float4* wr = (const float4*)(Wq + ((size_t)h * HD + tid) * QD);
        float acc = bq[h * HD + tid];
        #pragma unroll 4
        for (int i = 0; i < QD / 4; ++i) {
            float4 w4 = wr[i], q4 = qp[i];
            acc += w4.x * q4.x + w4.y * q4.y + w4.z * q4.z + w4.w * q4.w;
        }
        sm_wq[tid] = acc;
    } else {
        const int d = tid - HD;
        sm_bw[d] = bw[h * HD + d];
        sm_wv[d] = Wv[h * HD + d];
        sm_c[d]  = 0.0f;
    }
    if (tid == 0) { sm_ml[0] = -__builtin_inff(); sm_ml[1] = 0.0f; }

    // =========================== stream over S ===========================
    for (int st = 0; st < SS / TS; ++st) {
        const int tilebase = st * TS;

        // ---- GEMM1: head_kv tile = kv_tile (64x1024) x Ww[h]^T, K chunked by 128 ----
        v8f hk[4] = {};
        for (int kc = 0; kc < KVD; kc += 128) {
            __syncthreads();
            // stage kv chunk fp32->bf16 (coalesced float4 reads)
            {
                const float* src = kv + ((size_t)b * SS + tilebase) * KVD + kc;
                for (int c = tid; c < (TS * 128) / 4; c += 256) {
                    const int row = c >> 5;
                    const int kq  = (c & 31) << 2;
                    const float4 v = *(const float4*)(src + (size_t)row * KVD + kq);
                    v4bf o; o[0] = (__bf16)v.x; o[1] = (__bf16)v.y;
                            o[2] = (__bf16)v.z; o[3] = (__bf16)v.w;
                    *(v4bf*)(smAB + row * LDP + kq) = o;
                    // speculative prefetch of next k-chunk (global_prefetch_b8)
                    __builtin_prefetch(src + (size_t)row * KVD + kq + 128, 0, 0);
                }
            }
            // stage Ww chunk (bf16, contiguous along k)
            {
                const __bf16* wsrc = wsrcBase + kc;
                for (int c = tid; c < (HD * 128) / 8; c += 256) {
                    const int row = c >> 4;
                    const int ko  = (c & 15) << 3;
                    *(v8bf*)(smW + row * LDP + ko) =
                        *(const v8bf*)(wsrc + (size_t)row * KVD + ko);
                }
            }
            __syncthreads();
            #pragma unroll
            for (int kb = 0; kb < 128; kb += 32) {
                const v16bf a = ld_frag_a(smAB + arow * LDP, kb, grp);
                #pragma unroll
                for (int nt = 0; nt < 4; ++nt) {
                    const v16bf bfr = ld_frag_b(smW + (cb + nt * 16 + ln) * LDP, kb, grp);
                    hk[nt] = __builtin_amdgcn_wmma_f32_16x16x32_bf16(
                        false, a, false, bfr, (short)0, hk[nt], false, false);
                }
            }
        }
        __syncthreads();

        // ---- bias add; stash head_kv tile (bf16) as GEMM2 A; stage Wc; zero scores ----
        #pragma unroll
        for (int nt = 0; nt < 4; ++nt) {
            #pragma unroll
            for (int i = 0; i < 8; ++i) {
                const int col = cb + nt * 16 + ln;
                const int row = mb * 16 + i + grp * 8;     // C/D layout: M = i + 8*grp
                const float v = hk[nt][i] + sm_bw[col];
                hk[nt][i] = v;                              // keep f32 for ctx
                smAB[row * LDP + col] = (__bf16)v;
            }
        }
        if (tid < TS) sm_score[tid] = 0.0f;
        for (int c = tid; c < (HD * HD) / 8; c += 256) {
            const int row = c >> 4;
            const int ko  = (c & 15) << 3;
            *(v8bf*)(smW + row * LDP + ko) = *(const v8bf*)(csrc + row * HD + ko);
        }
        __syncthreads();

        // ---- GEMM2: uh = head_kv_tile (64x128) x Wc[h]^T ----
        v8f uh[4] = {};
        #pragma unroll
        for (int kb = 0; kb < HD; kb += 32) {
            const v16bf a = ld_frag_a(smAB + arow * LDP, kb, grp);
            #pragma unroll
            for (int nt = 0; nt < 4; ++nt) {
                const v16bf bfr = ld_frag_b(smW + (cb + nt * 16 + ln) * LDP, kb, grp);
                uh[nt] = __builtin_amdgcn_wmma_f32_16x16x32_bf16(
                    false, a, false, bfr, (short)0, uh[nt], false, false);
            }
        }

        // ---- scores: sum_d Wv[d] * tanh(wq[d] + uh[s,d]) ----
        #pragma unroll
        for (int nt = 0; nt < 4; ++nt) {
            #pragma unroll
            for (int i = 0; i < 8; ++i) {
                const int col = cb + nt * 16 + ln;
                const int row = mb * 16 + i + grp * 8;
                atomicAdd(&sm_score[row], sm_wv[col] * tanhf(sm_wq[col] + uh[nt][i]));
            }
        }
        __syncthreads();

        // ---- finalize scores (bias + mask), persist to workspace ----
        if (tid < TS) {
            float sv = sm_score[tid] + bv_h;
            if (maskp[tilebase + tid] != 0) sv = -__builtin_inff();
            sm_score[tid] = sv;
            scoresW[(size_t)bh * SS + tilebase + tid] = sv;
        }
        __syncthreads();

        // ---- online softmax: new max + rescale factor ----
        if (tid == 0) {
            float tm = -__builtin_inff();
            for (int r = 0; r < TS; ++r) tm = fmaxf(tm, sm_score[r]);
            const float mo = sm_ml[0];
            const float nm = fmaxf(fmaxf(mo, tm), -1.0e30f);
            sm_ml[2] = nm;
            sm_ml[3] = expf(mo - nm);
            sm_ml[0] = nm;
        }
        __syncthreads();
        if (tid < TS) sm_score[tid] = expf(sm_score[tid] - sm_ml[2]);  // p
        if (tid >= 128) sm_c[tid - 128] *= sm_ml[3];                    // rescale ctx
        __syncthreads();
        if (tid == 0) {
            float ssum = 0.0f;
            for (int r = 0; r < TS; ++r) ssum += sm_score[r];
            sm_ml[1] = sm_ml[1] * sm_ml[3] + ssum;
        }
        // ---- ctx accumulation from register-resident f32 head_kv ----
        #pragma unroll
        for (int nt = 0; nt < 4; ++nt) {
            #pragma unroll
            for (int i = 0; i < 8; ++i) {
                const int col = cb + nt * 16 + ln;
                const int row = mb * 16 + i + grp * 8;
                atomicAdd(&sm_c[col], sm_score[row] * hk[nt][i]);
            }
        }
        // top-of-loop __syncthreads orders these adds before buffer reuse
    }
    __syncthreads();
    if (tid < HD) outCtx[(size_t)b * (NH * HD) + h * HD + tid] = sm_c[tid] / sm_ml[1];
    if (tid == 0) { mW[bh] = sm_ml[0]; lW[bh] = sm_ml[1]; }
}

// ---------------------------------------------------------------------------
// Kernel 2: attn_max[b,s] = max_h exp(score - m_h) / l_h  (masked -> 0)
// ---------------------------------------------------------------------------
__global__ __launch_bounds__(256)
void attnmax_kernel(const float* __restrict__ scoresW,
                    const float* __restrict__ mW,
                    const float* __restrict__ lW,
                    float* __restrict__ outAmax) {
    const int idx = blockIdx.x * blockDim.x + threadIdx.x;   // B*S threads exactly
    const int b = idx / SS;
    const int s = idx - b * SS;
    float best = 0.0f;
    #pragma unroll
    for (int h = 0; h < NH; ++h) {
        const int bh = b * NH + h;
        const float a = expf(scoresW[(size_t)bh * SS + s] - mW[bh]) / lW[bh];
        best = fmaxf(best, a);
    }
    outAmax[idx] = best;
}

// ---------------------------------------------------------------------------
extern "C" void kernel_launch(void* const* d_in, const int* in_sizes, int n_in,
                              void* d_out, int out_size, void* d_ws, size_t ws_size,
                              hipStream_t stream) {
    (void)in_sizes; (void)n_in; (void)out_size; (void)ws_size;
    const float* q    = (const float*)d_in[0];
    const float* kv   = (const float*)d_in[1];
    const int*   mask = (const int*)  d_in[2];
    const float* Ww   = (const float*)d_in[3];
    const float* bw   = (const float*)d_in[4];
    const float* Wc   = (const float*)d_in[5];
    const float* Wq   = (const float*)d_in[6];
    const float* bq   = (const float*)d_in[7];
    const float* Wv   = (const float*)d_in[8];
    const float* bv   = (const float*)d_in[9];

    float* outCtx  = (float*)d_out;                       // [B, H*D] = [32,1024]
    float* outAmax = outCtx + (size_t)BB * NH * HD;       // [B, S]

    // workspace layout
    float* scoresW = (float*)d_ws;                        // B*H*S floats (2 MB)
    float* mW      = scoresW + (size_t)BB * NH * SS;
    float* lW      = mW + BB * NH;
    unsigned short* WwB =
        (unsigned short*)((char*)d_ws + ((size_t)BB * NH * SS + 2 * BB * NH) * sizeof(float));
    unsigned short* WcB = WwB + (size_t)NH * HD * KVD;

    prepack_kernel<<<4608, 256, 0, stream>>>(Ww, Wc, WwB, WcB);
    fused_attn_kernel<<<BB * NH, 256, 0, stream>>>(q, kv, mask, bw, Wq, bq, Wv, bv,
                                                   WwB, WcB, scoresW, mW, lW, outCtx);
    attnmax_kernel<<<(BB * SS) / 256, 256, 0, stream>>>(scoresW, mW, lW, outAmax);
}
